// MPCActor_28424093565697
// MI455X (gfx1250) — compile-verified, matrix operand order checked by reading, not run on hardware
//
#include <hip/hip_runtime.h>
#include <hip/hip_bf16.h>
#include <math.h>

typedef __attribute__((ext_vector_type(16))) _Float16 v16h;
typedef __attribute__((ext_vector_type(4)))  _Float16 v4h;
typedef __attribute__((ext_vector_type(8)))  float    v8f;
typedef unsigned int       u32;
typedef unsigned long long u64;

#define BATCH    32768
#define OBS_LEN  256
#define HIDDEN   512
#define OUT_DIM  32
#define LR       0.01f
#define N_ITERS  100

// ---------------------------------------------------------------------------
// Pack weights W (K x N, f32 row-major) into WMMA-fragment order, f16:
//   block[kt][nt] (kt = k/32, nt = n/16) = [lane 0..31][e 0..15] halves
//   lane = (n&15) + 16*((k>>3)&1)   (wave half handles K offset +8)
//   e    = (k&7)  +  8*((k>>4)&1)   (VGPRs 0..3 = K 0..7, VGPRs 4..7 = K 16..23)
// A wave then loads one B fragment as two contiguous b128 loads per lane.
// ---------------------------------------------------------------------------
__global__ __launch_bounds__(256) void pack_b_kernel(const float* __restrict__ src,
                                                     _Float16* __restrict__ dst,
                                                     int K, int N) {
  int i = blockIdx.x * blockDim.x + threadIdx.x;
  if (i >= K * N) return;
  int k = i / N, n = i % N;
  int kt = k >> 5, kl = k & 31, nt = n >> 4, nl = n & 15;
  int lane = nl + 16 * ((kl >> 3) & 1);
  int e    = (kl & 7) + 8 * ((kl >> 4) & 1);
  dst[((size_t)(kt * (N >> 4) + nt) * 32 + lane) * 16 + e] = (_Float16)src[i];
}

// ---------------------------------------------------------------------------
// WMMA GEMM. 256 threads = 8 waves; each wave owns a 32x32 tile (2x2 16x16).
// A staged in LDS in fragment-swizzled layout [mtile][lane][16 halves],
// double-buffered; f16 path fills it with global_load_async_to_lds_b128.
// B read directly from packed global (L2-resident weights).
// ACT==0: relu -> f16 out.  ACT==1: sigmoid -> f32 out.
// ---------------------------------------------------------------------------
template<int WM, int WN, int ACT, bool A_IS_F32>
__global__ __launch_bounds__(256) void gemm_wmma_kernel(
    const void* __restrict__ Aptr, const _Float16* __restrict__ Bp,
    const float* __restrict__ bias, void* __restrict__ Yout,
    int N, int K)
{
  constexpr int BK    = 32;
  constexpr int BM    = WM * 32;
  constexpr int BUF_H = BM * 32;          // halves per LDS buffer

  __shared__ __align__(16) _Float16 As[2 * BUF_H];

  const int tid    = threadIdx.x;
  const int lane   = tid & 31;
  const int wave   = tid >> 5;
  const int wave_m = wave / WN;
  const int wave_n = wave % WN;
  const int moff16 = wave_m * 2;          // first m-tile (of 16 rows) for this wave
  const int hh     = lane >> 4;
  const int mr     = lane & 15;

  const int row0   = blockIdx.y * BM;
  const int col0   = blockIdx.x * (WN * 32);
  const int ntile0 = (col0 >> 4) + wave_n * 2;
  const int ntiles = N >> 4;

  const u32 lds_base = (u32)(uintptr_t)(&As[0]);   // LDS byte offset of As

  v8f acc[2][2] = {};

  // ---- stage one BK-chunk of A into LDS buffer `buf`, swizzled ----
  auto stage = [&](int buf, int k0) {
    if constexpr (A_IS_F32) {
      const float* A = (const float*)Aptr;
      // unit = 4 halves (8B LDS store), global float4 load + cvt
      for (int ci = tid; ci < BM * 8; ci += 256) {
        int r = ci >> 3, q = ci & 7;
        int k = q * 4;
        int h = (k >> 3) & 1, g = (k >> 4) & 1, e0 = (k & 7) + 8 * g;
        const float4 f = *(const float4*)(A + (size_t)(row0 + r) * K + k0 + k);
        v4h hv;
        hv[0] = (_Float16)f.x; hv[1] = (_Float16)f.y;
        hv[2] = (_Float16)f.z; hv[3] = (_Float16)f.w;
        int off = buf * BUF_H + (((r >> 4) * 32) + (r & 15) + 16 * h) * 16 + e0;
        *(v4h*)&As[off] = hv;
      }
    } else {
      const _Float16* A = (const _Float16*)Aptr;
      // unit = 8 halves = 16B: async DMA straight into swizzled LDS slot
      for (int ci = tid; ci < BM * 4; ci += 256) {
        int r = ci >> 2, c = ci & 3;
        int h = c & 1, g = c >> 1;
        u32 loff = (u32)(buf * BUF_H +
                         (((r >> 4) * 32) + (r & 15) + 16 * h) * 16 + g * 8) * 2u;
        const _Float16* src = A + (size_t)(row0 + r) * K + k0 + c * 8;
        asm volatile("global_load_async_to_lds_b128 %0, %1, off"
                     :: "v"(lds_base + loff), "v"((u64)(uintptr_t)src)
                     : "memory");
      }
    }
  };

  // ---- compute one BK-chunk from LDS buffer `buf` ----
  auto compute = [&](int buf, int kblk) {
    v16h af[2];
    #pragma unroll
    for (int i = 0; i < 2; ++i) {
      int off = buf * BUF_H + ((moff16 + i) * 32 + lane) * 16;
      uint4* d = (uint4*)&af[i];
      d[0] = *(const uint4*)&As[off];       // ds_load_b128
      d[1] = *(const uint4*)&As[off + 8];   // ds_load_b128
    }
    v16h bf[2];
    #pragma unroll
    for (int j = 0; j < 2; ++j) {
      const uint4* s = (const uint4*)Bp +
                       ((size_t)kblk * ntiles + ntile0 + j) * 64 + lane * 2;
      uint4* d = (uint4*)&bf[j];
      d[0] = s[0];                          // global_load_b128 (L2-hot weights)
      d[1] = s[1];
    }
    #pragma unroll
    for (int i = 0; i < 2; ++i)
      #pragma unroll
      for (int j = 0; j < 2; ++j)
        acc[i][j] = __builtin_amdgcn_wmma_f32_16x16x32_f16(
            false, af[i], false, bf[j], (short)0, acc[i][j], false, false);
  };

  // ---- main loop: double-buffered async staging overlapped with WMMA ----
  int cur = 0;
  stage(0, 0);
  for (int k0 = 0; k0 < K; k0 += BK) {
    if constexpr (!A_IS_F32)
      asm volatile("s_wait_asynccnt 0x0" ::: "memory");
    __syncthreads();                        // staged chunk visible to all waves
    if (k0 + BK < K) stage(cur ^ 1, k0 + BK);
    compute(cur, k0 >> 5);
    __syncthreads();                        // reads done before buffer reuse
    cur ^= 1;
  }

  // ---- epilogue: bias + activation ----
  #pragma unroll
  for (int i = 0; i < 2; ++i) {
    #pragma unroll
    for (int j = 0; j < 2; ++j) {
      const int gn = col0 + wave_n * 32 + j * 16 + mr;
      const float bv = bias[gn];
      #pragma unroll
      for (int r = 0; r < 8; ++r) {
        const int gm = row0 + wave_m * 32 + i * 16 + r + 8 * hh;
        float v = acc[i][j][r] + bv;
        if constexpr (ACT == 0) {
          v = v > 0.0f ? v : 0.0f;
          ((_Float16*)Yout)[(size_t)gm * N + gn] = (_Float16)v;
        } else {
          v = 1.0f / (1.0f + __expf(-v));
          ((float*)Yout)[(size_t)gm * N + gn] = v;
        }
      }
    }
  }
}

// ---------------------------------------------------------------- GD iterations
__global__ __launch_bounds__(256) void iterate_kernel(const float* __restrict__ y3,
                                                      const float* __restrict__ u_init,
                                                      float* __restrict__ out, int n) {
  int idx = blockIdx.x * blockDim.x + threadIdx.x;
  if (idx >= n) return;
  int row = idx >> 2, j = idx & 3;
  float q = y3[row * OUT_DIM + 12 + j];   // q_u: cols 12..15
  float p = y3[row * OUT_DIM + 28 + j];   // p_u: cols 28..31
  float u = u_init[idx];
  const float a = 1.0f - 2.0f * LR * q;   // u <- a*u - LR*p
  #pragma unroll 10
  for (int t = 0; t < N_ITERS; ++t)
    u = a * u - LR * p;
  out[idx] = u;
}

// ---------------------------------------------------------------- launch
extern "C" void kernel_launch(void* const* d_in, const int* in_sizes, int n_in,
                              void* d_out, int out_size, void* d_ws, size_t ws_size,
                              hipStream_t stream) {
  const float* obs    = (const float*)d_in[0];
  // d_in[1] = x_init (unused by reference)
  const float* u_init = (const float*)d_in[2];
  const float* W1     = (const float*)d_in[3];
  const float* b1     = (const float*)d_in[4];
  const float* W2     = (const float*)d_in[5];
  const float* b2     = (const float*)d_in[6];
  const float* W3     = (const float*)d_in[7];
  const float* b3     = (const float*)d_in[8];
  float* out = (float*)d_out;

  // workspace layout
  char* ws = (char*)d_ws;
  _Float16* W1p = (_Float16*)ws;  ws += (size_t)OBS_LEN * HIDDEN  * 2;  // 256K
  _Float16* W2p = (_Float16*)ws;  ws += (size_t)HIDDEN  * HIDDEN  * 2;  // 512K
  _Float16* W3p = (_Float16*)ws;  ws += (size_t)HIDDEN  * OUT_DIM * 2;  // 32K
  float*    Y3  = (float*)ws;     ws += (size_t)BATCH   * OUT_DIM * 4;  // 4M
  _Float16* Y1  = (_Float16*)ws;  ws += (size_t)BATCH   * HIDDEN  * 2;  // 32M
  _Float16* Y2  = (_Float16*)ws;  ws += (size_t)BATCH   * HIDDEN  * 2;  // 32M

  // pack weights into WMMA fragment order (f32 -> f16)
  pack_b_kernel<<<(OBS_LEN * HIDDEN + 255) / 256, 256, 0, stream>>>(W1, W1p, OBS_LEN, HIDDEN);
  pack_b_kernel<<<(HIDDEN * HIDDEN + 255) / 256, 256, 0, stream>>>(W2, W2p, HIDDEN, HIDDEN);
  pack_b_kernel<<<(HIDDEN * OUT_DIM + 255) / 256, 256, 0, stream>>>(W3, W3p, HIDDEN, OUT_DIM);

  // layer 1: Y1 = relu(obs @ W1 + b1)   (fp32 A: load+cvt staging path)
  gemm_wmma_kernel<4, 2, 0, true>
      <<<dim3(HIDDEN / 64, BATCH / 128), 256, 0, stream>>>(
          obs, W1p, b1, Y1, HIDDEN, OBS_LEN);

  // layer 2: Y2 = relu(Y1 @ W2 + b2)    (async-to-LDS staging path)
  gemm_wmma_kernel<4, 2, 0, false>
      <<<dim3(HIDDEN / 64, BATCH / 128), 256, 0, stream>>>(
          Y1, W2p, b2, Y2, HIDDEN, HIDDEN);

  // layer 3: Y3 = sigmoid(Y2 @ W3 + b3) (BM=256, BN=32)
  gemm_wmma_kernel<8, 1, 1, false>
      <<<dim3(OUT_DIM / 32, BATCH / 256), 256, 0, stream>>>(
          Y2, W3p, b3, Y3, OUT_DIM, HIDDEN);

  // 100 GD steps on u
  const int n = BATCH * 4;
  iterate_kernel<<<(n + 255) / 256, 256, 0, stream>>>(Y3, u_init, out, n);
}
